// FreqEmbedding_34918084116490
// MI455X (gfx1250) — matrix operand order repaired and checked by reading, not compiled, not run on hardware
//
#include <hip/hip_runtime.h>
#include <math.h>

typedef _Float16 f16;
typedef __attribute__((ext_vector_type(16))) _Float16 v16h;
typedef __attribute__((ext_vector_type(8)))  float    v8f;

#define B_   16
#define L_   512
#define C_   64
#define W_   128      // window length (K of the GEMM)
#define F_   65       // rfft bins
#define NF   80       // freq bins padded to 5 tiles of 16
#define PADT 64       // reflection pad
#define TW   640      // padded signal length
#define NTOT (L_ * F_)

// B fragment (32x16 f16): lane = column f, element e -> K = k0 + g*16 + e
// basis stored transposed [f][w] so the 16 halves are contiguous (32B aligned).
__device__ __forceinline__ v16h load_bfrag(const f16* bas, int f, int k0, int g) {
    return *(const v16h*)(bas + f * W_ + k0 + g * 16);
}

// A fragment (16x32 f16): lane group g: element e -> K = k0 + g*16 + e.
// A[row, K] = p[row + K]  (sliding window), so element e reads p[t0 + e].
__device__ __forceinline__ v16h load_afrag(const f16* p, int t0) {
    v16h a;
#pragma unroll
    for (int e = 0; e < 8; ++e) {
        a[e]     = p[t0 + e];
        a[e + 8] = p[t0 + 8 + e];
    }
    return a;
}

__global__ __launch_bounds__(256) void freq_embed_kernel(const float* __restrict__ seq,
                                                         float* __restrict__ out) {
    const int b    = blockIdx.x >> 6;   // / C_
    const int c    = blockIdx.x & 63;   // % C_
    const int tid  = threadIdx.x;
    const int lane = tid & 31;
    const int wave = tid >> 5;
    const int r    = lane & 15;         // column within tile for B/C/D, row for A
    const int g    = lane >> 4;         // lane half-group

    __shared__ __align__(32) f16 pH[TW], pL[TW];                 // hi/lo split signal
    __shared__ __align__(32) f16 brH[NF * W_], brL[NF * W_];     // windowed DFT real, [f][w]
    __shared__ __align__(32) f16 biH[NF * W_], biL[NF * W_];     // windowed DFT imag, [f][w]
    __shared__ float red[8][3];
    __shared__ float stats[3];

    // ---- reflection-padded signal, split into f16 hi + lo residual ----
    for (int t = tid; t < TW; t += 256) {
        int i   = t - PADT;
        int src = (i < 0) ? -i : ((i >= L_) ? (2 * (L_ - 1) - i) : i);
        float v = seq[((size_t)b * L_ + src) * C_ + c];
        f16 hi  = (f16)v;
        pH[t]   = hi;
        pL[t]   = (f16)(v - (float)hi);
    }

    // ---- windowed DFT basis, transposed [f][w], zero for f >= 65 ----
    const float twopi = 6.283185307179586f;
    for (int idx = tid; idx < NF * W_; idx += 256) {
        int f = idx >> 7;    // / 128
        int w = idx & 127;
        float re = 0.0f, im = 0.0f;
        if (f < F_) {
            float h  = 0.5f - 0.5f * cosf(twopi * (float)w / 127.0f); // hann, len-1 = 127
            int   ph = (f * w) & 127;                                  // exact mod-128 phase
            float th = (twopi / 128.0f) * (float)ph;
            re =  h * cosf(th);
            im = -h * sinf(th);   // rfft: e^{-i theta}
        }
        f16 reh  = (f16)re;
        f16 imh  = (f16)im;
        brH[idx] = reh;  brL[idx] = (f16)(re - (float)reh);
        biH[idx] = imh;  biL[idx] = (f16)(im - (float)imh);
    }
    __syncthreads();

    // ---- one 16x16 output tile pair (re & im) : K = 128 in 4 steps of 32 ----
    // D = A_hi*B_hi + A_hi*B_lo + A_lo*B_hi  (error-compensated f16 GEMM)
    auto tile = [&](int m0, int f0, v8f& accR, v8f& accI) {
        accR = {};
        accI = {};
        const int fcol = f0 + r;
#pragma unroll
        for (int s = 0; s < 4; ++s) {
            const int k0 = s * 32;
            const int t0 = m0 + r + k0 + g * 16;
            v16h aH  = load_afrag(pH, t0);
            v16h aL  = load_afrag(pL, t0);
            v16h bRH = load_bfrag(brH, fcol, k0, g);
            v16h bRL = load_bfrag(brL, fcol, k0, g);
            v16h bIH = load_bfrag(biH, fcol, k0, g);
            v16h bIL = load_bfrag(biL, fcol, k0, g);
            accR = __builtin_amdgcn_wmma_f32_16x16x32_f16(false, aH, false, bRH, (short)0, accR, false, false);
            accR = __builtin_amdgcn_wmma_f32_16x16x32_f16(false, aH, false, bRL, (short)0, accR, false, false);
            accR = __builtin_amdgcn_wmma_f32_16x16x32_f16(false, aL, false, bRH, (short)0, accR, false, false);
            accI = __builtin_amdgcn_wmma_f32_16x16x32_f16(false, aH, false, bIH, (short)0, accI, false, false);
            accI = __builtin_amdgcn_wmma_f32_16x16x32_f16(false, aH, false, bIL, (short)0, accI, false, false);
            accI = __builtin_amdgcn_wmma_f32_16x16x32_f16(false, aL, false, bIH, (short)0, accI, false, false);
        }
    };

    // ---- sweep 1: accumulate sum(re), sum(im), sum(|z|^2) ----
    // (f in [65,80) columns are exactly zero -> contribute nothing; no mask needed)
    float sR = 0.0f, sI = 0.0f, s2 = 0.0f;
    for (int tp = wave; tp < 32 * 5; tp += 8) {
        const int m0 = (tp / 5) * 16;
        const int f0 = (tp % 5) * 16;
        v8f aR, aI;
        tile(m0, f0, aR, aI);
#pragma unroll
        for (int v = 0; v < 8; ++v) {
            sR += aR[v];
            sI += aI[v];
            s2 += aR[v] * aR[v] + aI[v] * aI[v];
        }
    }
#pragma unroll
    for (int off = 16; off > 0; off >>= 1) {
        sR += __shfl_xor(sR, off, 32);
        sI += __shfl_xor(sI, off, 32);
        s2 += __shfl_xor(s2, off, 32);
    }
    if (lane == 0) { red[wave][0] = sR; red[wave][1] = sI; red[wave][2] = s2; }
    __syncthreads();
    if (tid == 0) {
        float SR = 0.0f, SI = 0.0f, S2 = 0.0f;
        for (int i = 0; i < 8; ++i) { SR += red[i][0]; SI += red[i][1]; S2 += red[i][2]; }
        const float invN = 1.0f / (float)NTOT;
        float muR = SR * invN;
        float muI = SI * invN;
        float var = S2 * invN - muR * muR - muI * muI;   // E|z-mu|^2 = E|z|^2 - |mu|^2
        stats[0] = muR; stats[1] = muI; stats[2] = rsqrtf(var);
    }
    __syncthreads();
    const float muR = stats[0], muI = stats[1], inv = stats[2];

    // ---- sweep 2: recompute (LDS-resident, compute is ~free) and write |z-mu|/std ----
    for (int tp = wave; tp < 32 * 5; tp += 8) {
        const int m0 = (tp / 5) * 16;
        const int f0 = (tp % 5) * 16;
        v8f aR, aI;
        tile(m0, f0, aR, aI);
        const int fcol = f0 + r;
        if (fcol < F_) {
#pragma unroll
            for (int v = 0; v < 8; ++v) {
                const int m = m0 + g * 8 + v;           // D: VGPR v -> row v + 8*g
                float zr = aR[v] - muR;
                float zi = aI[v] - muI;
                out[(((size_t)b * L_ + m) * F_ + fcol) * C_ + c] = sqrtf(zr * zr + zi * zi) * inv;
            }
        }
    }
}

extern "C" void kernel_launch(void* const* d_in, const int* in_sizes, int n_in,
                              void* d_out, int out_size, void* d_ws, size_t ws_size,
                              hipStream_t stream) {
    (void)in_sizes; (void)n_in; (void)out_size; (void)d_ws; (void)ws_size;
    const float* seq = (const float*)d_in[0];
    float* out = (float*)d_out;
    freq_embed_kernel<<<dim3(B_ * C_), dim3(256), 0, stream>>>(seq, out);
}
